// AR_Back_Step_23467701305809
// MI455X (gfx1250) — matrix operand order repaired and checked by reading, not compiled, not run on hardware
//
#include <hip/hip_runtime.h>
#include <hip/hip_bf16.h>
#include <math.h>

// ---------------------------------------------------------------------------
// RADTTS-style AR decoder for MI455X (gfx1250, wave32, WMMA f16->f32).
// Strategy:
//   * Hoist all time-parallel GEMMs (input-gate precompute, attention
//     projections, decoder MLP) into v_wmma_f32_16x16x32_f16 tiled GEMMs with
//     f16 weights pre-packed into exact WMMA B-fragment order (32B/lane loads).
//   * The unavoidable serial LSTM recurrences run in a single 1024-thread
//     workgroup per layer: per step, 32 waves compute the (4x1024)x(1024x4096)
//     GEMV via WMMA streaming W_hh (f16, resident in 192MB L2), gates go to
//     LDS, the f32 LSTM cell updates per-thread (h,c) carries.
//   * Fused wave-per-row additive-attention kernel (scores/softmax/context/
//     logprob/gates) — VALU-bound tanh work.
// ---------------------------------------------------------------------------

typedef __attribute__((ext_vector_type(16))) _Float16 v16h;
typedef __attribute__((ext_vector_type(8)))  float    v8f;

#define T_MELN 800
#define NB     4
#define NMEL   80
#define T_TXTN 256
#define TXTD   640
#define HID    1024
#define AD     128
#define G4     4096            // 4*HID gate width
#define ROWS   (T_MELN * NB)   // 3200

// output layout (flat f32, in reference return order)
#define MELOUT_OFF 0
#define LOGS_OFF   256000
#define GATES_OFF  512000
#define ATTN_OFF   515200
#define LP_OFF     1334400

union AFrag {
  v16h v;
  unsigned u[8];
  _Float16 h[16];
};

__device__ __forceinline__ float sigm(float x) { return 1.0f / (1.0f + __expf(-x)); }

__device__ __forceinline__ float wave_sum(float s) {
#pragma unroll
  for (int o = 16; o > 0; o >>= 1) s += __shfl_xor(s, o, 32);
  return s;
}
__device__ __forceinline__ float wave_max(float s) {
#pragma unroll
  for (int o = 16; o > 0; o >>= 1) s = fmaxf(s, __shfl_xor(s, o, 32));
  return s;
}

// ---------------------------------------------------------------------------
// Pack W (N x K, f32, row-major, we compute X @ W^T) into f16 WMMA B-fragment
// order: frag (nt,kt) -> 32 lanes x 16 halves; lane holds column n = lane&15,
// halves are K = kt*32 + (lane>=16 ? 8 : 0) + (h<8 ? h : h+8)   (ISA B-matrix striping).
// One thread writes one lane's 32B chunk.
// ---------------------------------------------------------------------------
__global__ void pack_wT(const float* __restrict__ W, int N, int K, int KT,
                        int total, _Float16* __restrict__ out) {
  int gid = blockIdx.x * blockDim.x + threadIdx.x;
  if (gid >= total) return;                 // total = NT*KT*32
  int lane = gid & 31;
  int frag = gid >> 5;
  int kt = frag % KT;
  int nt = frag / KT;
  int n  = nt * 16 + (lane & 15);
  int kb = kt * 32 + ((lane >= 16) ? 8 : 0);
  AFrag f;
#pragma unroll
  for (int h = 0; h < 16; ++h) {
    int k = kb + (h < 8 ? h : h + 8);
    float v = (n < N && k < K) ? W[(size_t)n * K + k] : 0.0f;
    f.h[h] = (_Float16)v;
  }
  *(v16h*)(out + (size_t)gid * 16) = f.v;
}

// ---------------------------------------------------------------------------
// Generic WMMA GEMM: C(M,N) = act(A(M,Kpad) @ Bpk + bias).  One wave per
// 16x16 C tile; A gathered per ISA A-fragment striping (b32 pair loads),
// B loaded as one contiguous v16h per lane from the packed buffer.
// ---------------------------------------------------------------------------
__global__ void wmma_gemm(const _Float16* __restrict__ A, int kpad, int KT,
                          const _Float16* __restrict__ Bpk,
                          const float* __restrict__ bias,
                          float* __restrict__ Cf, _Float16* __restrict__ Ch,
                          int ldc, int act) {
  const int mt = blockIdx.x, nt = blockIdx.y;
  const int lane = threadIdx.x;
  const bool hi = lane >= 16;
  const int mrow = mt * 16 + (lane & 15);
  const _Float16* arow = A + (size_t)mrow * kpad + (hi ? 8 : 0);
  const _Float16* bbase = Bpk + (size_t)nt * KT * 32 * 16;
  v8f acc = {};
  for (int kt = 0; kt < KT; ++kt) {
    AFrag a;
    const _Float16* ap = arow + kt * 32;
    a.u[0] = *(const unsigned*)(ap + 0);
    a.u[1] = *(const unsigned*)(ap + 2);
    a.u[2] = *(const unsigned*)(ap + 4);
    a.u[3] = *(const unsigned*)(ap + 6);
    a.u[4] = *(const unsigned*)(ap + 16);
    a.u[5] = *(const unsigned*)(ap + 18);
    a.u[6] = *(const unsigned*)(ap + 20);
    a.u[7] = *(const unsigned*)(ap + 22);
    v16h b = *(const v16h*)(bbase + ((size_t)kt * 32 + lane) * 16);
    acc = __builtin_amdgcn_wmma_f32_16x16x32_f16(false, a.v, false, b,
                                                 (short)0, acc, false, false);
  }
  const int col = nt * 16 + (lane & 15);
  const float bv = bias ? bias[col] : 0.0f;
#pragma unroll
  for (int r = 0; r < 8; ++r) {
    int row = mt * 16 + r + (hi ? 8 : 0);
    float vv = acc[r] + bv;
    if (act) vv = tanhf(vv);
    if (Cf) Cf[(size_t)row * ldc + col] = vv;
    if (Ch) Ch[(size_t)row * ldc + col] = (_Float16)vv;
  }
}

// ---------------------------------------------------------------------------
// Serial LSTM recurrence, one layer.  Single 1024-thread workgroup (32 waves,
// 8 j-tiles each).  X holds precomputed x@W_ih^T + (b_ih+b_hh) for all steps.
// Per step: WMMA GEMV h(f16,LDS) x W_hh(f16, packed, streamed from L2),
// gates -> LDS (f16), then f32 cell update with per-thread (h,c) carries and
// the t<len masking (carry frozen, output zeroed).
// ---------------------------------------------------------------------------
__global__ __launch_bounds__(1024) void lstm_rec(
    const float* __restrict__ X, const _Float16* __restrict__ Wpk,
    const int* __restrict__ lens, _Float16* __restrict__ hout) {
  __shared__ __align__(16) _Float16 gbuf[4 * G4];   // 32 KB
  __shared__ __align__(16) _Float16 h16[4 * HID];   // 8 KB
  const int tid = threadIdx.x;
  const int lane = tid & 31, wave = tid >> 5;
  const bool hi = lane >= 16;
  const int mrow = lane & 15;
  const int mm = mrow & 3;
  const unsigned zmask = (mrow < 4) ? 0xFFFFFFFFu : 0u;
  for (int i = tid; i < 4 * HID; i += 1024) h16[i] = (_Float16)0;
  float c[4]  = {0, 0, 0, 0};
  float hc[4] = {0, 0, 0, 0};
  int Ls[4];
#pragma unroll
  for (int b = 0; b < 4; ++b) Ls[b] = lens[b];
  const int u = tid;  // hidden unit owned by this thread
  __syncthreads();

  for (int t = 0; t < T_MELN; ++t) {
    // ---- recurrent GEMV via WMMA ----
    for (int jj = 0; jj < 8; ++jj) {
      const int jt = wave * 8 + jj;
      const _Float16* bp = Wpk + (size_t)jt * 16384 + (size_t)lane * 16;
      const _Float16* hbase = h16 + mm * HID + (hi ? 8 : 0);
      v8f acc = {};
      for (int kt = 0; kt < 32; ++kt) {
        AFrag a;
        const _Float16* hp = hbase + kt * 32;
        a.u[0] = *(const unsigned*)(hp + 0)  & zmask;
        a.u[1] = *(const unsigned*)(hp + 2)  & zmask;
        a.u[2] = *(const unsigned*)(hp + 4)  & zmask;
        a.u[3] = *(const unsigned*)(hp + 6)  & zmask;
        a.u[4] = *(const unsigned*)(hp + 16) & zmask;
        a.u[5] = *(const unsigned*)(hp + 18) & zmask;
        a.u[6] = *(const unsigned*)(hp + 20) & zmask;
        a.u[7] = *(const unsigned*)(hp + 22) & zmask;
        if (kt + 1 < 32) __builtin_prefetch(bp + (size_t)(kt + 1) * 512, 0, 0);
        v16h b = *(const v16h*)(bp + (size_t)kt * 512);
        acc = __builtin_amdgcn_wmma_f32_16x16x32_f16(false, a.v, false, b,
                                                     (short)0, acc, false, false);
      }
      if (lane < 16) {  // rows 0..3 (batch) live in acc[0..3] on lanes 0..15
        int j = jt * 16 + lane;
        gbuf[0 * G4 + j] = (_Float16)acc[0];
        gbuf[1 * G4 + j] = (_Float16)acc[1];
        gbuf[2 * G4 + j] = (_Float16)acc[2];
        gbuf[3 * G4 + j] = (_Float16)acc[3];
      }
    }
    __syncthreads();
    // ---- f32 cell update ----
#pragma unroll
    for (int b = 0; b < 4; ++b) {
      const float* xr = X + (size_t)(t * 4 + b) * G4;
      float gi = (float)gbuf[b * G4 + u]            + xr[u];
      float gf = (float)gbuf[b * G4 + HID + u]      + xr[HID + u];
      float gg = (float)gbuf[b * G4 + 2 * HID + u]  + xr[2 * HID + u];
      float go = (float)gbuf[b * G4 + 3 * HID + u]  + xr[3 * HID + u];
      float cn = sigm(gf) * c[b] + sigm(gi) * tanhf(gg);
      float hn = sigm(go) * tanhf(cn);
      bool valid = t < Ls[b];
      c[b]  = valid ? cn : c[b];
      hc[b] = valid ? hn : hc[b];
      hout[(size_t)(t * 4 + b) * HID + u] = valid ? (_Float16)hn : (_Float16)0;
    }
    __syncthreads();
#pragma unroll
    for (int b = 0; b < 4; ++b) h16[b * HID + u] = (_Float16)hc[b];
    __syncthreads();
  }
}

// ---------------------------------------------------------------------------
// Fused additive attention: one wave per (m,b) row.  scores -> masked softmax
// -> attn/logprob (written in the final rolled frame) -> context -> dec_in
// assembly -> Wg gate dot.
// ---------------------------------------------------------------------------
__global__ void attn_ctx(const float* __restrict__ Q, const float* __restrict__ Kf,
                         const float* __restrict__ Vf, const float* __restrict__ vain,
                         const unsigned char* __restrict__ mask,
                         const int* __restrict__ lens,
                         const _Float16* __restrict__ attnh,
                         const float* __restrict__ Wg, const float* __restrict__ bg,
                         _Float16* __restrict__ dec_in, float* __restrict__ out) {
  const int lane = threadIdx.x & 31;
  const int wid = blockIdx.x * (blockDim.x >> 5) + (threadIdx.x >> 5);  // 0..3199
  const int m = wid >> 2, b = wid & 3;
  float qv[4], va[4];
#pragma unroll
  for (int i = 0; i < 4; ++i) {
    qv[i] = Q[(size_t)wid * AD + lane + 32 * i];
    va[i] = vain[lane + 32 * i];
  }
  float sc[8];
#pragma unroll
  for (int i = 0; i < 8; ++i) {
    for (int j = 0; j < 32; ++j) {
      int tt = i * 32 + j;
      const float* kr = Kf + (size_t)(tt * 4 + b) * AD;
      float s = 0.0f;
#pragma unroll
      for (int q = 0; q < 4; ++q) s += va[q] * tanhf(qv[q] + kr[lane + 32 * q]);
      s = wave_sum(s);
      if (mask[b * T_TXTN + tt]) s = -INFINITY;
      if (lane == j) sc[i] = s;
    }
  }
  float mx = sc[0];
#pragma unroll
  for (int i = 1; i < 8; ++i) mx = fmaxf(mx, sc[i]);
  mx = wave_max(mx);
  float ssum = 0.0f;
#pragma unroll
  for (int i = 0; i < 8; ++i) { sc[i] = __expf(sc[i] - mx); ssum += sc[i]; }
  ssum = wave_sum(ssum);
  const float inv = 1.0f / ssum;
  const int mp = (m - lens[b] + T_MELN) % T_MELN;  // roll by -out_lens
  float* ao = out + (size_t)ATTN_OFF + ((size_t)b * T_MELN + mp) * T_TXTN;
  float* lo = out + (size_t)LP_OFF + ((size_t)b * T_MELN + mp) * T_TXTN;
#pragma unroll
  for (int i = 0; i < 8; ++i) {
    int tt = i * 32 + lane;
    float p = sc[i] * inv;
    ao[tt] = p;
    lo[tt] = __logf(p + 1e-8f);
    sc[i] = p;
  }
  float ctx[4] = {0, 0, 0, 0};
  for (int i = 0; i < 8; ++i) {
    for (int j = 0; j < 32; ++j) {
      float p = __shfl(sc[i], j, 32);
      const float* vr = Vf + (size_t)((i * 32 + j) * 4 + b) * AD;
#pragma unroll
      for (int q = 0; q < 4; ++q) ctx[q] += p * vr[lane + 32 * q];
    }
  }
  _Float16* dr = dec_in + (size_t)wid * (HID + AD);
  const _Float16* ar = attnh + (size_t)wid * HID;
  float g = 0.0f;
  for (int j2 = lane; j2 < HID; j2 += 32) {
    _Float16 hv = ar[j2];
    dr[j2] = hv;
    g += (float)hv * Wg[j2];
  }
#pragma unroll
  for (int q = 0; q < 4; ++q) {
    int a = lane + 32 * q;
    dr[HID + a] = (_Float16)ctx[q];
    g += ctx[q] * Wg[HID + a];
  }
  g = wave_sum(g);
  if (lane == 0) out[GATES_OFF + wid] = g + bg[0];
}

// ---------------------------------------------------------------------------
// Small prep / epilogue kernels
// ---------------------------------------------------------------------------
__global__ void prep_mel0(const float* __restrict__ mel, const int* __restrict__ lens,
                          _Float16* __restrict__ m0) {
  int idx = blockIdx.x * blockDim.x + threadIdx.x;
  if (idx >= ROWS * 96) return;
  int k = idx % 96;
  int r = idx / 96;
  int b = r & 3;
  int t = r >> 2;
  float v = 0.0f;
  if (k < NMEL && t > 0) {
    int src = (T_MELN - t + lens[b]) % T_MELN;  // flip+roll frame of step t-1
    v = mel[(size_t)src * (NB * NMEL) + b * NMEL + k];
  }
  m0[idx] = (_Float16)v;
}

__global__ void f32_to_f16(const float* __restrict__ in, _Float16* __restrict__ out, int n) {
  int i = blockIdx.x * blockDim.x + threadIdx.x;
  if (i < n) out[i] = (_Float16)in[i];
}

__global__ void add_bias2(const float* __restrict__ a, const float* __restrict__ b,
                          float* __restrict__ o, int n) {
  int i = blockIdx.x * blockDim.x + threadIdx.x;
  if (i < n) o[i] = a[i] + b[i];
}

__global__ void final_mel(const float* __restrict__ dec_out, const float* __restrict__ mel,
                          const int* __restrict__ lens, float* __restrict__ out) {
  int idx = blockIdx.x * blockDim.x + threadIdx.x;
  if (idx >= ROWS * NMEL) return;
  int k = idx % NMEL;
  int r = idx / NMEL;
  int b = r & 3;
  int t = r >> 2;
  float ls = dec_out[(size_t)r * 160 + k];
  float bb = dec_out[(size_t)r * 160 + 80 + k];
  int src = (T_MELN - 1 - t + lens[b]) % T_MELN;  // involution: work <-> final
  float mv = mel[(size_t)src * (NB * NMEL) + b * NMEL + k];
  out[MELOUT_OFF + (size_t)src * (NB * NMEL) + b * NMEL + k] = __expf(ls) * mv + bb;
  out[LOGS_OFF + (size_t)t * (NB * NMEL) + b * NMEL + k] = ls;
}

// ---------------------------------------------------------------------------
extern "C" void kernel_launch(void* const* d_in, const int* in_sizes, int n_in,
                              void* d_out, int out_size, void* d_ws, size_t ws_size,
                              hipStream_t stream) {
  (void)in_sizes; (void)n_in; (void)out_size; (void)ws_size;
  // inputs (setup_inputs dict order, params flattened in insertion order)
  const float* mel    = (const float*)d_in[0];
  const float* text   = (const float*)d_in[1];
  const unsigned char* mask = (const unsigned char*)d_in[2];
  const int*   lens   = (const int*)d_in[3];
  const float* aWih   = (const float*)d_in[4];   // (4096,80)
  const float* aWhh   = (const float*)d_in[5];   // (4096,1024)
  const float* aBih   = (const float*)d_in[6];
  const float* aBhh   = (const float*)d_in[7];
  const float* l0Wih  = (const float*)d_in[8];   // (4096,1152)
  const float* l0Whh  = (const float*)d_in[9];
  const float* l0Bih  = (const float*)d_in[10];
  const float* l0Bhh  = (const float*)d_in[11];
  const float* l1Wih  = (const float*)d_in[12];  // (4096,1024)
  const float* l1Whh  = (const float*)d_in[13];
  const float* l1Bih  = (const float*)d_in[14];
  const float* l1Bhh  = (const float*)d_in[15];
  const float* Wq     = (const float*)d_in[16];  // (128,1024)
  const float* Wk     = (const float*)d_in[17];  // (128,640)
  const float* Wv     = (const float*)d_in[18];
  const float* vattn  = (const float*)d_in[19];  // (128,)
  const float* Wg     = (const float*)d_in[20];  // (1,1152)
  const float* bg     = (const float*)d_in[21];
  const float* Wd1    = (const float*)d_in[22];  // (1024,1024)
  const float* bd1    = (const float*)d_in[23];
  const float* Wd2    = (const float*)d_in[24];
  const float* bd2    = (const float*)d_in[25];
  const float* Wc     = (const float*)d_in[26];  // (160,1024)
  const float* bc     = (const float*)d_in[27];
  float* out = (float*)d_out;

  // workspace carve-out
  char* w = (char*)d_ws;
  size_t off = 0;
  auto alloc = [&](size_t bytes) -> void* {
    off = (off + 255) & ~(size_t)255;
    void* p = w + off;
    off += bytes;
    return p;
  };
  auto hsz = [](size_t halves) { return halves * sizeof(_Float16); };

  // packed f16 weights (B-fragment order): frags = NT*KT, 512 halves each
  _Float16* Bih0 = (_Float16*)alloc(hsz((size_t)256 * 3 * 512));
  _Float16* Whh0 = (_Float16*)alloc(hsz((size_t)256 * 32 * 512));
  _Float16* Bih1 = (_Float16*)alloc(hsz((size_t)256 * 36 * 512));
  _Float16* Whh1 = (_Float16*)alloc(hsz((size_t)256 * 32 * 512));
  _Float16* Bih2 = (_Float16*)alloc(hsz((size_t)256 * 32 * 512));
  _Float16* Whh2 = (_Float16*)alloc(hsz((size_t)256 * 32 * 512));
  _Float16* BWq  = (_Float16*)alloc(hsz((size_t)8 * 32 * 512));
  _Float16* BWk  = (_Float16*)alloc(hsz((size_t)8 * 20 * 512));
  _Float16* BWv  = (_Float16*)alloc(hsz((size_t)8 * 20 * 512));
  _Float16* BWd1 = (_Float16*)alloc(hsz((size_t)64 * 32 * 512));
  _Float16* BWd2 = (_Float16*)alloc(hsz((size_t)64 * 32 * 512));
  _Float16* BWc  = (_Float16*)alloc(hsz((size_t)10 * 32 * 512));
  float* bias0 = (float*)alloc(G4 * sizeof(float));
  float* bias1 = (float*)alloc(G4 * sizeof(float));
  float* bias2 = (float*)alloc(G4 * sizeof(float));
  // f16 activations (A-operand row-major)
  _Float16* mel0h  = (_Float16*)alloc(hsz((size_t)ROWS * 96));
  _Float16* texth  = (_Float16*)alloc(hsz((size_t)T_TXTN * NB * TXTD));
  _Float16* attnh  = (_Float16*)alloc(hsz((size_t)ROWS * HID));
  _Float16* decin  = (_Float16*)alloc(hsz((size_t)ROWS * (HID + AD)));
  _Float16* h1h    = (_Float16*)alloc(hsz((size_t)ROWS * HID));
  _Float16* h2h    = (_Float16*)alloc(hsz((size_t)ROWS * HID));
  _Float16* d1h    = (_Float16*)alloc(hsz((size_t)ROWS * HID));
  _Float16* d2h    = (_Float16*)alloc(hsz((size_t)ROWS * HID));
  // f32 scratch
  float* X     = (float*)alloc((size_t)ROWS * G4 * sizeof(float));  // reused 3x
  float* keysf = (float*)alloc((size_t)T_TXTN * NB * AD * sizeof(float));
  float* valsf = (float*)alloc((size_t)T_TXTN * NB * AD * sizeof(float));
  float* qf    = (float*)alloc((size_t)ROWS * AD * sizeof(float));
  float* decof = (float*)alloc((size_t)ROWS * 160 * sizeof(float));

  auto packs = [&](const float* W_, int N, int K, int KT, _Float16* dst) {
    int total = (N / 16) * KT * 32;
    pack_wT<<<(total + 255) / 256, 256, 0, stream>>>(W_, N, K, KT, total, dst);
  };
  // weight packing (f16 fragment layouts)
  packs(aWih,  4096,   80,  3, Bih0);
  packs(aWhh,  4096, 1024, 32, Whh0);
  packs(l0Wih, 4096, 1152, 36, Bih1);
  packs(l0Whh, 4096, 1024, 32, Whh1);
  packs(l1Wih, 4096, 1024, 32, Bih2);
  packs(l1Whh, 4096, 1024, 32, Whh2);
  packs(Wq,     128, 1024, 32, BWq);
  packs(Wk,     128,  640, 20, BWk);
  packs(Wv,     128,  640, 20, BWv);
  packs(Wd1,   1024, 1024, 32, BWd1);
  packs(Wd2,   1024, 1024, 32, BWd2);
  packs(Wc,     160, 1024, 32, BWc);
  add_bias2<<<16, 256, 0, stream>>>(aBih, aBhh, bias0, G4);
  add_bias2<<<16, 256, 0, stream>>>(l0Bih, l0Bhh, bias1, G4);
  add_bias2<<<16, 256, 0, stream>>>(l1Bih, l1Bhh, bias2, G4);
  prep_mel0<<<(ROWS * 96 + 255) / 256, 256, 0, stream>>>(mel, lens, mel0h);
  f32_to_f16<<<(T_TXTN * NB * TXTD + 255) / 256, 256, 0, stream>>>(text, texth, T_TXTN * NB * TXTD);

  // attention LSTM: X0 = mel0 @ aWih^T + bias0, then serial recurrence
  wmma_gemm<<<dim3(ROWS / 16, 256), 32, 0, stream>>>(mel0h, 96, 3, Bih0, bias0, X, nullptr, G4, 0);
  lstm_rec<<<1, 1024, 0, stream>>>(X, Whh0, lens, attnh);

  // attention projections
  wmma_gemm<<<dim3(64, 8), 32, 0, stream>>>(texth, TXTD, 20, BWk, nullptr, keysf, nullptr, AD, 0);
  wmma_gemm<<<dim3(64, 8), 32, 0, stream>>>(texth, TXTD, 20, BWv, nullptr, valsf, nullptr, AD, 0);
  wmma_gemm<<<dim3(ROWS / 16, 8), 32, 0, stream>>>(attnh, HID, 32, BWq, nullptr, qf, nullptr, AD, 0);

  // fused scores/softmax/context/logprob/gates + dec_in assembly
  attn_ctx<<<ROWS / 8, 256, 0, stream>>>(qf, keysf, valsf, vattn, mask, lens,
                                         attnh, Wg, bg, decin, out);

  // decoder LSTM layer 1
  wmma_gemm<<<dim3(ROWS / 16, 256), 32, 0, stream>>>(decin, HID + AD, 36, Bih1, bias1, X, nullptr, G4, 0);
  lstm_rec<<<1, 1024, 0, stream>>>(X, Whh1, lens, h1h);
  // decoder LSTM layer 2
  wmma_gemm<<<dim3(ROWS / 16, 256), 32, 0, stream>>>(h1h, HID, 32, Bih2, bias2, X, nullptr, G4, 0);
  lstm_rec<<<1, 1024, 0, stream>>>(X, Whh2, lens, h2h);

  // decoder MLP + head
  wmma_gemm<<<dim3(ROWS / 16, 64), 32, 0, stream>>>(h2h, HID, 32, BWd1, bd1, nullptr, d1h, HID, 1);
  wmma_gemm<<<dim3(ROWS / 16, 64), 32, 0, stream>>>(d1h, HID, 32, BWd2, bd2, nullptr, d2h, HID, 1);
  wmma_gemm<<<dim3(ROWS / 16, 10), 32, 0, stream>>>(d2h, HID, 32, BWc, bc, decof, nullptr, 160, 0);

  // mel_out / log_s with un-roll/un-flip applied at write time
  final_mel<<<(ROWS * NMEL + 255) / 256, 256, 0, stream>>>(decof, mel, lens, out);
}
